// PropagationLayer_35905926595017
// MI455X (gfx1250) — matrix operand order repaired and checked by reading, not compiled, not run on hardware
//
#include <hip/hip_runtime.h>
#include <math.h>

typedef __attribute__((ext_vector_type(16))) _Float16 v16h;
typedef __attribute__((ext_vector_type(8)))  float    v8f;

#define IN_F   256
#define OUT_F  64
#define NHEAD  4
#define PROJ_W 128   // fwd(64) | rev(64) concatenated per node

// ---- order-preserving float <-> uint for atomicMax on signed floats ----
__device__ __forceinline__ unsigned encode_ordered(float f) {
  unsigned u = __float_as_uint(f);
  return (u & 0x80000000u) ? ~u : (u | 0x80000000u);
}
__device__ __forceinline__ float decode_ordered(unsigned k) {
  unsigned u = (k & 0x80000000u) ? (k ^ 0x80000000u) : ~k;
  return __uint_as_float(u);
}

// ============================================================================
// Projection GEMM: hproj[n, 0:64] = h[n,:] @ Wf.T ; hproj[n, 64:128] = h @ Wr.T
// One block = 16 rows; 8 waves cover the 128 output columns (16 each).
// One wave computes a 16x16 f32 tile via v_wmma_f32_16x16x32_f16, K-loop of 8.
// ============================================================================
__global__ void __launch_bounds__(256) gat_proj_gemm(
    const float* __restrict__ h, const float* __restrict__ Wf,
    const float* __restrict__ Wr, float* __restrict__ hproj, int N) {
  const int lane = threadIdx.x & 31;
  const int wave = threadIdx.x >> 5;       // 0..7 -> column tile
  const int col0 = wave * 16;
  const int row0 = blockIdx.x * 16;
  const int nrow = lane & 15;              // A: row M ; B/D: column N
  const int hi   = lane >> 4;              // lane half

  const int arow = (row0 + nrow < N) ? (row0 + nrow) : (N - 1);
  const float* __restrict__ ha = h + (size_t)arow * IN_F;

  const int j = col0 + nrow;               // output column 0..127
  const float* __restrict__ wsrc = (j < OUT_F)
      ? (Wf + (size_t)j * IN_F)
      : (Wr + (size_t)(j - OUT_F) * IN_F);

  v8f acc = {};
  for (int kb = 0; kb < IN_F; kb += 32) {
    // ---- A fragment: lane half hi holds K = kb+8*hi + {0..7} and +16 ----
    const int ka = kb + hi * 8;
    const float4 a0 = *(const float4*)(ha + ka);
    const float4 a1 = *(const float4*)(ha + ka + 4);
    const float4 a2 = *(const float4*)(ha + ka + 16);
    const float4 a3 = *(const float4*)(ha + ka + 20);
    v16h A;
    A[0]=(_Float16)a0.x;  A[1]=(_Float16)a0.y;  A[2]=(_Float16)a0.z;  A[3]=(_Float16)a0.w;
    A[4]=(_Float16)a1.x;  A[5]=(_Float16)a1.y;  A[6]=(_Float16)a1.z;  A[7]=(_Float16)a1.w;
    A[8]=(_Float16)a2.x;  A[9]=(_Float16)a2.y;  A[10]=(_Float16)a2.z; A[11]=(_Float16)a2.w;
    A[12]=(_Float16)a3.x; A[13]=(_Float16)a3.y; A[14]=(_Float16)a3.z; A[15]=(_Float16)a3.w;

    // ---- B fragment: lane half hi holds K = kb+8*hi + {0..7} and +16 ----
    const int kbv = kb + hi * 8;
    const float4 b0 = *(const float4*)(wsrc + kbv);
    const float4 b1 = *(const float4*)(wsrc + kbv + 4);
    const float4 b2 = *(const float4*)(wsrc + kbv + 16);
    const float4 b3 = *(const float4*)(wsrc + kbv + 20);
    v16h B;
    B[0]=(_Float16)b0.x;  B[1]=(_Float16)b0.y;  B[2]=(_Float16)b0.z;  B[3]=(_Float16)b0.w;
    B[4]=(_Float16)b1.x;  B[5]=(_Float16)b1.y;  B[6]=(_Float16)b1.z;  B[7]=(_Float16)b1.w;
    B[8]=(_Float16)b2.x;  B[9]=(_Float16)b2.y;  B[10]=(_Float16)b2.z; B[11]=(_Float16)b2.w;
    B[12]=(_Float16)b3.x; B[13]=(_Float16)b3.y; B[14]=(_Float16)b3.z; B[15]=(_Float16)b3.w;

    acc = __builtin_amdgcn_wmma_f32_16x16x32_f16(
        /*neg_a=*/false, A, /*neg_b=*/false, B,
        /*c_mod=*/(short)0, acc, /*reuse_a=*/false, /*reuse_b=*/false);
  }

  // Epilogue: fast unguarded path for full tiles (the common case), guarded
  // path only for a ragged final row tile. VGPR r holds row M = r + 8*hi.
  if (row0 + 16 <= N) {
    float* __restrict__ dcol =
        hproj + (size_t)(row0 + hi * 8) * PROJ_W + col0 + nrow;
    #pragma unroll
    for (int r = 0; r < 8; ++r) dcol[(size_t)r * PROJ_W] = acc[r];
  } else {
    #pragma unroll
    for (int r = 0; r < 8; ++r) {
      const int gr = row0 + r + hi * 8;
      if (gr < N) hproj[(size_t)gr * PROJ_W + col0 + nrow] = acc[r];
    }
  }
}

// ============================================================================
// Edge core: one warp per edge. Lane l owns features 2l, 2l+1 of the message.
// Produces message pair (m0,m1), leaky-relu'd per-head scores s[4]
// (valid on ALL lanes after butterfly reduce), and dst.
// ============================================================================
__device__ __forceinline__ void edge_core(
    int e, int E, int lane,
    const long long* __restrict__ adj, const float* __restrict__ dir,
    const float* __restrict__ hproj,  const float* __restrict__ attn,
    float& m0, float& m1, float (&s)[NHEAD], long long& dst) {
  const long long src = adj[e];
  dst = adj[(size_t)E + e];
  const float d0 = dir[2 * (size_t)e];
  const float d1 = dir[2 * (size_t)e + 1];
  const int f = lane * 2;
  const float* __restrict__ hp = hproj + (size_t)src * PROJ_W;
  const float2 hf = *(const float2*)(hp + f);
  const float2 hr = *(const float2*)(hp + OUT_F + f);
  m0 = hf.x * d0 + hr.x * d1;
  m1 = hf.y * d0 + hr.y * d1;
  #pragma unroll
  for (int hd = 0; hd < NHEAD; ++hd) {
    float p = m0 * attn[hd * OUT_F + f] + m1 * attn[hd * OUT_F + f + 1];
    #pragma unroll
    for (int off = 16; off > 0; off >>= 1) p += __shfl_xor(p, off, 32);
    s[hd] = (p > 0.0f) ? p : 0.2f * p;     // leaky_relu(0.2)
  }
}

// ---- Pass 1: global per-head max (grid-stride, block-reduced atomics) ----
__global__ void __launch_bounds__(256) gat_edge_max(
    const long long* __restrict__ adj, const float* __restrict__ dir,
    const float* __restrict__ hproj,  const float* __restrict__ attn,
    unsigned* __restrict__ head_max, int E) {
  const int lane = threadIdx.x & 31;
  const int warp = threadIdx.x >> 5;
  const int nwarps = (gridDim.x * blockDim.x) >> 5;
  float mx[NHEAD] = {-INFINITY, -INFINITY, -INFINITY, -INFINITY};
  for (int e = (blockIdx.x * blockDim.x + threadIdx.x) >> 5; e < E; e += nwarps) {
    float m0, m1, s[NHEAD]; long long dst;
    edge_core(e, E, lane, adj, dir, hproj, attn, m0, m1, s, dst);
    #pragma unroll
    for (int hd = 0; hd < NHEAD; ++hd) mx[hd] = fmaxf(mx[hd], s[hd]);
  }
  __shared__ float sh[8][NHEAD];
  if (lane == 0)
    for (int hd = 0; hd < NHEAD; ++hd) sh[warp][hd] = mx[hd];
  __syncthreads();
  if (threadIdx.x < NHEAD) {
    float v = sh[0][threadIdx.x];
    #pragma unroll
    for (int w = 1; w < 8; ++w) v = fmaxf(v, sh[w][threadIdx.x]);
    atomicMax(&head_max[threadIdx.x], encode_ordered(v));
  }
}

// ---- Pass 2: sum_exp[dst, hd] += exp(s - max) ----
__global__ void __launch_bounds__(256) gat_edge_sumexp(
    const long long* __restrict__ adj, const float* __restrict__ dir,
    const float* __restrict__ hproj,  const float* __restrict__ attn,
    const unsigned* __restrict__ head_max, float* __restrict__ sum_exp, int E) {
  const int e = blockIdx.x * 8 + (threadIdx.x >> 5);
  if (e >= E) return;
  const int lane = threadIdx.x & 31;
  float m0, m1, s[NHEAD]; long long dst;
  edge_core(e, E, lane, adj, dir, hproj, attn, m0, m1, s, dst);
  if (lane < NHEAD) {
    const float mx = decode_ordered(head_max[lane]);
    atomicAdd(&sum_exp[dst * NHEAD + lane], __expf(s[lane] - mx));
  }
}

// ---- Pass 3: out[dst, hd*64+f] += message[f] * exp(s-max)/(sum_exp+eps) ----
__global__ void __launch_bounds__(256) gat_edge_scatter(
    const long long* __restrict__ adj, const float* __restrict__ dir,
    const float* __restrict__ hproj,  const float* __restrict__ attn,
    const unsigned* __restrict__ head_max, const float* __restrict__ sum_exp,
    float* __restrict__ out, int E) {
  const int e = blockIdx.x * 8 + (threadIdx.x >> 5);
  if (e >= E) return;
  const int lane = threadIdx.x & 31;
  float m0, m1, s[NHEAD]; long long dst;
  edge_core(e, E, lane, adj, dir, hproj, attn, m0, m1, s, dst);
  float nrm[NHEAD];
  #pragma unroll
  for (int hd = 0; hd < NHEAD; ++hd) {
    const float mx = decode_ordered(head_max[hd]);
    nrm[hd] = __expf(s[hd] - mx) / (sum_exp[dst * NHEAD + hd] + 1e-10f);
  }
  const int f = lane * 2;
  float* __restrict__ orow = out + (size_t)dst * (NHEAD * OUT_F);
  #pragma unroll
  for (int hd = 0; hd < NHEAD; ++hd) {
    atomicAdd(orow + hd * OUT_F + f,     m0 * nrm[hd]);
    atomicAdd(orow + hd * OUT_F + f + 1, m1 * nrm[hd]);
  }
}

// ============================================================================
extern "C" void kernel_launch(void* const* d_in, const int* in_sizes, int n_in,
                              void* d_out, int out_size, void* d_ws, size_t ws_size,
                              hipStream_t stream) {
  const float*     h    = (const float*)d_in[0];
  const long long* adj  = (const long long*)d_in[1];   // int64 [2, E]
  const float*     dir  = (const float*)d_in[2];       // [E, 2]
  const float*     Wf   = (const float*)d_in[3];       // [64, 256]
  const float*     Wr   = (const float*)d_in[4];       // [64, 256]
  const float*     attn = (const float*)d_in[5];       // [1, 4, 64] flat
  float* out = (float*)d_out;

  const int N = in_sizes[0] / IN_F;
  const int E = in_sizes[1] / 2;

  // workspace: hproj [N,128] f32 | sum_exp [N,4] f32 | head_max [4] u32
  float*    hproj    = (float*)d_ws;
  float*    sum_exp  = hproj + (size_t)N * PROJ_W;
  unsigned* head_max = (unsigned*)(sum_exp + (size_t)N * NHEAD);

  hipMemsetAsync(d_out, 0, (size_t)out_size * sizeof(float), stream);
  hipMemsetAsync(sum_exp, 0, (size_t)N * NHEAD * sizeof(float), stream);
  hipMemsetAsync(head_max, 0, NHEAD * sizeof(unsigned), stream);

  gat_proj_gemm<<<(N + 15) / 16, 256, 0, stream>>>(h, Wf, Wr, hproj, N);

  gat_edge_max<<<1024, 256, 0, stream>>>(adj, dir, hproj, attn, head_max, E);

  const int eb = (E + 7) / 8;   // warp per edge, 8 warps/block
  gat_edge_sumexp<<<eb, 256, 0, stream>>>(adj, dir, hproj, attn, head_max, sum_exp, E);
  gat_edge_scatter<<<eb, 256, 0, stream>>>(adj, dir, hproj, attn, head_max, sum_exp, out, E);
}
